// AdvancedDeepSpeedCNN_52201032515811
// MI455X (gfx1250) — compile-verified, matrix operand order checked by reading, not run on hardware
//
#include <hip/hip_runtime.h>
#include <hip/hip_bf16.h>
#include <math.h>

typedef __attribute__((ext_vector_type(16))) __bf16 v16bf;
typedef __attribute__((ext_vector_type(8)))  float  v8f;

#define DIMC   1024
#define NHEADS 16
#define HDIM   64
#define SEQ    2048
#define BATCH  2
#define BHTOT  (BATCH * NHEADS)

__device__ __forceinline__ __bf16 tobf(float f) { return (__bf16)f; }

// Element K-index held by (vgpr i, packed slot j, lane-half khalf) for the
// 16-bit 16x32 A fragment (and symmetric B fragment) per CDNA5 ISA 7.12.2.
// Contiguous in groups of 8 -> compiler emits ds_load_b128 pairs.
__device__ __forceinline__ int kmap(int i, int j, int khalf) {
  return ((i & 4) << 2) + (khalf << 3) + ((i & 3) << 1) + j;
}

// Low 32 bits of a generic pointer to LDS == LDS byte address (flat->LDS
// aperture mapping truncates to addr[31:0]).
__device__ __forceinline__ unsigned ldsAddr(const void* p) {
  return (unsigned)(unsigned long long)p;
}

// ---------------------------------------------------------------------------
// Tiled bf16 WMMA GEMM:  out[m,n] = sum_k A[m,k] * W[n,k]   (both K-major)
// Block tile 128x64, 8 waves in a 4x2 grid, 32x32 per wave -> 4 WMMA/k-step.
// mode 0: scatter 3072-wide QKV output into bf16 q/k/v [B,H,N,D] buffers
// mode 1: f32 out[m*N+n] = acc + bias[n]
// ---------------------------------------------------------------------------
__global__ void __launch_bounds__(256)
gemm_wmma_bf16(const float* __restrict__ A, const float* __restrict__ W,
               int M, int N, int K,
               float* __restrict__ outF, const float* __restrict__ bias,
               __bf16* __restrict__ qb, __bf16* __restrict__ kb,
               __bf16* __restrict__ vb, int mode) {
  __shared__ __bf16 As[128][40];  // 128 rows x 32 k (padded)
  __shared__ __bf16 Bs[64][40];   // 64 cols  x 32 k (padded)

  const int tid   = threadIdx.x;
  const int wave  = tid >> 5;
  const int lane  = tid & 31;
  const int l16   = lane & 15;
  const int khalf = lane >> 4;
  const int mW    = (wave >> 1) * 32;   // wave row offset in block tile
  const int nW    = (wave & 1) * 32;    // wave col offset in block tile
  const int mBase = blockIdx.y * 128;
  const int nBase = blockIdx.x * 64;

  const int rb = tid >> 2;          // 0..63  B row (= output col)
  const int kb4 = (tid & 3) * 8;    // 0,8,16,24

  v8f acc00 = {}, acc01 = {}, acc10 = {}, acc11 = {};

  for (int k0 = 0; k0 < K; k0 += 32) {
    // Stage A: 128x32, two 8-element slots per thread
#pragma unroll
    for (int s = 0; s < 2; ++s) {
      const int slot = tid + s * 256;
      const int r  = slot >> 2;
      const int kk = (slot & 3) * 8;
      const float* ap = A + (size_t)(mBase + r) * K + k0 + kk;
#pragma unroll
      for (int u = 0; u < 8; ++u) As[r][kk + u] = tobf(ap[u]);
      if (k0 + 32 < K) __builtin_prefetch(ap + 32, 0, 3);
    }
    // Stage B: 64x32, one 8-element slot per thread
    {
      const float* bp = W + (size_t)(nBase + rb) * K + k0 + kb4;
#pragma unroll
      for (int u = 0; u < 8; ++u) Bs[rb][kb4 + u] = tobf(bp[u]);
      if (k0 + 32 < K) __builtin_prefetch(bp + 32, 0, 3);
    }
    __syncthreads();

    v16bf a0, a1, b0, b1;
#pragma unroll
    for (int i = 0; i < 8; ++i)
#pragma unroll
      for (int j = 0; j < 2; ++j) {
        const int kk = kmap(i, j, khalf);
        a0[2 * i + j] = As[mW + l16][kk];
        a1[2 * i + j] = As[mW + 16 + l16][kk];
        b0[2 * i + j] = Bs[nW + l16][kk];
        b1[2 * i + j] = Bs[nW + 16 + l16][kk];
      }
    acc00 = __builtin_amdgcn_wmma_f32_16x16x32_bf16(false, a0, false, b0,
                                                    (short)0, acc00, false, false);
    acc01 = __builtin_amdgcn_wmma_f32_16x16x32_bf16(false, a0, false, b1,
                                                    (short)0, acc01, false, false);
    acc10 = __builtin_amdgcn_wmma_f32_16x16x32_bf16(false, a1, false, b0,
                                                    (short)0, acc10, false, false);
    acc11 = __builtin_amdgcn_wmma_f32_16x16x32_bf16(false, a1, false, b1,
                                                    (short)0, acc11, false, false);
    __syncthreads();
  }

#pragma unroll
  for (int rt = 0; rt < 2; ++rt)
#pragma unroll
    for (int ct = 0; ct < 2; ++ct) {
      v8f accv = (rt == 0) ? (ct == 0 ? acc00 : acc01)
                           : (ct == 0 ? acc10 : acc11);
#pragma unroll
      for (int i = 0; i < 8; ++i) {
        const int m = mBase + mW + rt * 16 + i + (khalf << 3);
        const int n = nBase + nW + ct * 16 + l16;
        const float val = accv[i];
        if (mode == 1) {
          outF[(size_t)m * N + n] = val + bias[n];
        } else {
          const int which = n >> 10;        // 0=q 1=k 2=v
          const int h     = (n >> 6) & 15;
          const int d     = n & 63;
          const int b     = m >> 11;        // m / SEQ
          const int s     = m & (SEQ - 1);
          __bf16* dst = (which == 0) ? qb : (which == 1) ? kb : vb;
          dst[(((size_t)(b * NHEADS + h)) * SEQ + s) * HDIM + d] = tobf(val);
        }
      }
    }
}

// ---------------------------------------------------------------------------
// In-place RoPE on bf16 q,k [B*H, N, D]; pair (d, d+32), freq index d (d<32).
// ---------------------------------------------------------------------------
__global__ void __launch_bounds__(256)
rope_kernel(__bf16* __restrict__ qb, __bf16* __restrict__ kb) {
  const int idx = blockIdx.x * blockDim.x + threadIdx.x;  // BH*SEQ*32 threads
  const int d  = idx & 31;
  const int n  = (idx >> 5) & (SEQ - 1);
  const int bh = idx >> 16;
  const size_t base = (((size_t)bh * SEQ) + n) * HDIM;

  const float inv = __powf(10000.f, -(float)(2 * d) / (float)HDIM);
  float s, c;
  __sincosf((float)n * inv, &s, &c);

  float q0 = (float)qb[base + d], q1 = (float)qb[base + d + 32];
  qb[base + d]      = tobf(q0 * c - q1 * s);
  qb[base + d + 32] = tobf(q1 * c + q0 * s);
  float k0 = (float)kb[base + d], k1 = (float)kb[base + d + 32];
  kb[base + d]      = tobf(k0 * c - k1 * s);
  kb[base + d + 32] = tobf(k1 * c + k0 * s);
}

// ---------------------------------------------------------------------------
// Flash attention with ALiBi. Block = 128 q-rows of one (b,h); 8 waves x 16
// rows; K/V chunks of 32 keys staged straight into LDS with CDNA5 async
// loads (ASYNCcnt); online softmax; WMMA for S = Q K^T and O += P V.
// ---------------------------------------------------------------------------
__global__ void __launch_bounds__(256)
attn_kernel(const __bf16* __restrict__ qb, const __bf16* __restrict__ kb,
            const __bf16* __restrict__ vb, const float* __restrict__ slopes,
            float* __restrict__ out) {
  __shared__ __bf16 Ks[32][72];          // 32 keys x 64 dims (row pitch 144B)
  __shared__ __bf16 Vs[32][72];
  __shared__ __bf16 Ps[8][16][40];       // per-wave P tile 16x32 (padded)

  const int tid   = threadIdx.x;
  const int wave  = tid >> 5;
  const int lane  = tid & 31;
  const int l16   = lane & 15;
  const int khalf = lane >> 4;

  const int bh    = blockIdx.y;
  const int b     = bh >> 4;
  const int h     = bh & 15;
  const int qBase = blockIdx.x * 128;
  const float slope8 = slopes[h] * 8.0f;   // MAX_BIAS = 8
  const float scale  = 0.125f;             // HEAD_DIM^-0.5

  const size_t hdBase = (size_t)bh * SEQ * HDIM;

  // Q fragments (A layout) for this wave's 16 rows, both K-steps over D=64.
  const int qRow = qBase + wave * 16 + l16;
  v16bf aQ0, aQ1;
  {
    const __bf16* qp = qb + hdBase + (size_t)qRow * HDIM;
#pragma unroll
    for (int i = 0; i < 8; ++i)
#pragma unroll
      for (int j = 0; j < 2; ++j) {
        const int kk = kmap(i, j, khalf);
        aQ0[2 * i + j] = qp[kk];
        aQ1[2 * i + j] = qp[32 + kk];
      }
  }

  v8f o0 = {}, o1 = {}, o2 = {}, o3 = {};
  float mrow[8], lrow[8];
#pragma unroll
  for (int i = 0; i < 8; ++i) { mrow[i] = -1e30f; lrow[i] = 0.f; }

  // Async staging map: thread owns one 16B segment of K and of V per chunk.
  const int lr = tid >> 3;               // 0..31 key row
  const int lk = (tid & 7) * 8;          // 0..56 dim offset (16B aligned)
  const unsigned kLds = ldsAddr(&Ks[lr][lk]);
  const unsigned vLds = ldsAddr(&Vs[lr][lk]);

  for (int c = 0; c < SEQ; c += 32) {
    const unsigned long long kga =
        (unsigned long long)(kb + hdBase + (size_t)(c + lr) * HDIM + lk);
    const unsigned long long vga =
        (unsigned long long)(vb + hdBase + (size_t)(c + lr) * HDIM + lk);
    // CDNA5 direct global->LDS async copies (no VGPR bounce), ASYNCcnt.
    asm volatile("global_load_async_to_lds_b128 %0, %1, off"
                 :: "v"(kLds), "v"(kga) : "memory");
    asm volatile("global_load_async_to_lds_b128 %0, %1, off"
                 :: "v"(vLds), "v"(vga) : "memory");
    asm volatile("s_wait_asynccnt 0x0" ::: "memory");
    __syncthreads();

    // S tiles: keys [c, c+16) and [c+16, c+32)
    v8f s0 = {}, s1 = {};
    {
      v16bf bK;
#pragma unroll
      for (int i = 0; i < 8; ++i)
#pragma unroll
        for (int j = 0; j < 2; ++j) bK[2 * i + j] = Ks[l16][kmap(i, j, khalf)];
      s0 = __builtin_amdgcn_wmma_f32_16x16x32_bf16(false, aQ0, false, bK,
                                                   (short)0, s0, false, false);
#pragma unroll
      for (int i = 0; i < 8; ++i)
#pragma unroll
        for (int j = 0; j < 2; ++j) bK[2 * i + j] = Ks[l16][32 + kmap(i, j, khalf)];
      s0 = __builtin_amdgcn_wmma_f32_16x16x32_bf16(false, aQ1, false, bK,
                                                   (short)0, s0, false, false);
#pragma unroll
      for (int i = 0; i < 8; ++i)
#pragma unroll
        for (int j = 0; j < 2; ++j) bK[2 * i + j] = Ks[16 + l16][kmap(i, j, khalf)];
      s1 = __builtin_amdgcn_wmma_f32_16x16x32_bf16(false, aQ0, false, bK,
                                                   (short)0, s1, false, false);
#pragma unroll
      for (int i = 0; i < 8; ++i)
#pragma unroll
        for (int j = 0; j < 2; ++j) bK[2 * i + j] = Ks[16 + l16][32 + kmap(i, j, khalf)];
      s1 = __builtin_amdgcn_wmma_f32_16x16x32_bf16(false, aQ1, false, bK,
                                                   (short)0, s1, false, false);
    }

    // Online softmax. A row of the C layout spans the 16 lanes of one half,
    // so xor-shuffles with masks 1,2,4,8 reduce exactly over one row.
    const int j0 = c + l16;
    const int j1 = c + 16 + l16;
#pragma unroll
    for (int i = 0; i < 8; ++i) {
      const int qi = qBase + wave * 16 + i + (khalf << 3);
      float v0 = s0[i] * scale;
      float v1 = s1[i] * scale;
      float d0 = (float)(j0 - qi); if (d0 > 0.f) d0 = 0.f;
      float d1 = (float)(j1 - qi); if (d1 > 0.f) d1 = 0.f;
      v0 += slope8 * d0;
      v1 += slope8 * d1;

      float rm = fmaxf(v0, v1);
#pragma unroll
      for (int msk = 8; msk >= 1; msk >>= 1)
        rm = fmaxf(rm, __shfl_xor(rm, msk, 32));
      const float mnew = fmaxf(mrow[i], rm);
      const float sc = __expf(mrow[i] - mnew);
      const float p0 = __expf(v0 - mnew);
      const float p1 = __expf(v1 - mnew);
      float rs = p0 + p1;
#pragma unroll
      for (int msk = 8; msk >= 1; msk >>= 1)
        rs += __shfl_xor(rs, msk, 32);
      lrow[i] = lrow[i] * sc + rs;
      mrow[i] = mnew;
      o0[i] *= sc; o1[i] *= sc; o2[i] *= sc; o3[i] *= sc;

      // Stage P (C layout -> LDS); same-wave DS ops are in-order, no barrier.
      Ps[wave][i + (khalf << 3)][l16]      = tobf(p0);
      Ps[wave][i + (khalf << 3)][16 + l16] = tobf(p1);
    }

    // O += P(16x32) @ V(32x64)
    v16bf aP;
#pragma unroll
    for (int i = 0; i < 8; ++i)
#pragma unroll
      for (int j = 0; j < 2; ++j)
        aP[2 * i + j] = Ps[wave][l16][kmap(i, j, khalf)];
#pragma unroll
    for (int t = 0; t < 4; ++t) {
      v16bf bV;
#pragma unroll
      for (int i = 0; i < 8; ++i)
#pragma unroll
        for (int j = 0; j < 2; ++j)
          bV[2 * i + j] = Vs[kmap(i, j, khalf)][t * 16 + l16];
      v8f* op = (t == 0) ? &o0 : (t == 1) ? &o1 : (t == 2) ? &o2 : &o3;
      *op = __builtin_amdgcn_wmma_f32_16x16x32_bf16(false, aP, false, bV,
                                                    (short)0, *op, false, false);
    }
    __syncthreads();
  }

  // Normalize and write f32 [B, N, H*D]
#pragma unroll
  for (int t = 0; t < 4; ++t) {
    v8f ov = (t == 0) ? o0 : (t == 1) ? o1 : (t == 2) ? o2 : o3;
#pragma unroll
    for (int i = 0; i < 8; ++i) {
      const int qi = qBase + wave * 16 + i + (khalf << 3);
      const int d  = t * 16 + l16;
      out[(((size_t)b * SEQ) + qi) * DIMC + h * HDIM + d] = ov[i] / lrow[i];
    }
  }
}

// ---------------------------------------------------------------------------
extern "C" void kernel_launch(void* const* d_in, const int* in_sizes, int n_in,
                              void* d_out, int out_size, void* d_ws,
                              size_t ws_size, hipStream_t stream) {
  const float* x      = (const float*)d_in[0];
  const float* qkv_w  = (const float*)d_in[1];
  const float* proj_w = (const float*)d_in[2];
  const float* proj_b = (const float*)d_in[3];
  const float* slopes = (const float*)d_in[4];
  float* out = (float*)d_out;

  const size_t qkvElems = (size_t)BHTOT * SEQ * HDIM;  // 4M elems per tensor
  char* ws = (char*)d_ws;
  __bf16* qb = (__bf16*)ws;  ws += qkvElems * sizeof(__bf16);
  __bf16* kb = (__bf16*)ws;  ws += qkvElems * sizeof(__bf16);
  __bf16* vb = (__bf16*)ws;  ws += qkvElems * sizeof(__bf16);
  float* attn = (float*)ws;  // BATCH*SEQ*DIMC f32 (16 MB)

  const int M = BATCH * SEQ;   // 4096

  // 1) QKV projection -> bf16 q/k/v [B,H,N,D]
  dim3 g1((3 * DIMC) / 64, M / 128);
  gemm_wmma_bf16<<<g1, 256, 0, stream>>>(x, qkv_w, M, 3 * DIMC, DIMC,
                                         nullptr, nullptr, qb, kb, vb, 0);
  // 2) RoPE in-place on q,k
  rope_kernel<<<(BHTOT * SEQ * 32) / 256, 256, 0, stream>>>(qb, kb);
  // 3) Flash attention with ALiBi -> f32 [B,N,C]
  dim3 g3(SEQ / 128, BHTOT);
  attn_kernel<<<g3, 256, 0, stream>>>(qb, kb, vb, slopes, attn);
  // 4) Output projection + bias
  dim3 g4(DIMC / 64, M / 128);
  gemm_wmma_bf16<<<g4, 256, 0, stream>>>(attn, proj_w, M, DIMC, DIMC,
                                         out, proj_b, nullptr, nullptr,
                                         nullptr, 1);
}
